// EncoderBlock_14671608283230
// MI455X (gfx1250) — matrix operand order repaired and checked by reading, not compiled, not run on hardware
//
#include <hip/hip_runtime.h>
#include <math.h>

// ---------------------------------------------------------------------------
// MI455X (gfx1250) transformer encoder block.
// All matmuls use v_wmma_f32_16x16x32_bf16 (wave32 WMMA, fp32 accumulate).
// GEMMs stage tiles in LDS via async global->LDS copies (ASYNCcnt) with
// double buffering; flash-style attention with per-wave LDS P-relayout.
// ---------------------------------------------------------------------------

typedef __attribute__((ext_vector_type(16))) __bf16 bf16x16;
typedef __attribute__((ext_vector_type(8)))  __bf16 bf16x8;
typedef __attribute__((ext_vector_type(8)))  float  f32x8;
typedef __attribute__((ext_vector_type(4)))  int    i32x4;

#define AS1 __attribute__((address_space(1)))
#define AS3 __attribute__((address_space(3)))

#if defined(__has_builtin)
#  if __has_builtin(__builtin_amdgcn_global_load_async_to_lds_b128)
#    define USE_ASYNC_LDS 1
#  endif
#endif
#ifndef USE_ASYNC_LDS
#  define USE_ASYNC_LDS 0
#endif

union BFrag { bf16x16 v; bf16x8 h[2]; };

__device__ inline unsigned short f2bf_u(float f) {
    union { float f; unsigned u; } c; c.f = f;
    unsigned r = c.u + 0x7FFFu + ((c.u >> 16) & 1u);  // round-to-nearest-even
    return (unsigned short)(r >> 16);
}

__device__ inline bf16x8 load_bf8(const unsigned short* p) {
    bf16x8 r;
    __builtin_memcpy(&r, p, 16);   // -> global_load_b128 / ds_load_b128
    return r;
}

__device__ inline f32x8 wmma_bf16(const BFrag& a, const BFrag& b, f32x8 c) {
    return __builtin_amdgcn_wmma_f32_16x16x32_bf16(
        /*neg_a=*/false, a.v, /*neg_b=*/false, b.v,
        /*c_mod=*/(short)0, c, /*reuse_a=*/false, /*reuse_b=*/false);
}

// 16B global -> LDS copy, async (ASYNCcnt) when the gfx1250 builtin exists.
// Signature (from hipcc diagnostic): (global i32x4*, local i32x4*, imm, imm).
__device__ inline void copy16_g2l(const unsigned short* g, unsigned short* l) {
#if USE_ASYNC_LDS
    __builtin_amdgcn_global_load_async_to_lds_b128(
        (AS1 i32x4*)g, (AS3 i32x4*)l, 0, 0);
#else
    bf16x8 t;
    __builtin_memcpy(&t, g, 16);
    __builtin_memcpy(l, &t, 16);
#endif
}

__device__ inline void wait_async_copies() {
#if USE_ASYNC_LDS
#  if defined(__has_builtin) && __has_builtin(__builtin_amdgcn_s_wait_asynccnt)
    __builtin_amdgcn_s_wait_asynccnt(0);
#  else
    asm volatile("s_wait_asynccnt 0" ::: "memory");
#  endif
#endif
}

// ---------------------------------------------------------------------------
// fp32 -> bf16 convert (weights)
// ---------------------------------------------------------------------------
__global__ __launch_bounds__(256)
void cvt_f32_to_bf16_kernel(const float* __restrict__ in,
                            unsigned short* __restrict__ out, int n) {
    int i = blockIdx.x * 256 + threadIdx.x;
    if (i < n) out[i] = f2bf_u(in[i]);
}

// ---------------------------------------------------------------------------
// LayerNorm (torch-style: unbiased std ddof=1, eps added to std), D = 1024.
// One 256-thread block per row; output bf16 for the following WMMA GEMMs.
// ---------------------------------------------------------------------------
__global__ __launch_bounds__(256)
void layernorm_bf16_kernel(const float* __restrict__ x,
                           const float* __restrict__ gamma,
                           const float* __restrict__ beta,
                           unsigned short* __restrict__ out) {
    __shared__ float s_sum[256];
    __shared__ float s_sq[256];
    const int tid = threadIdx.x;
    const int row = blockIdx.x;
    const float* xr = x + (size_t)row * 1024;
    float v[4]; float s = 0.f, sq = 0.f;
    for (int i = 0; i < 4; ++i) {
        v[i] = xr[tid + 256 * i];
        s += v[i]; sq += v[i] * v[i];
    }
    s_sum[tid] = s; s_sq[tid] = sq;
    __syncthreads();
    for (int st = 128; st > 0; st >>= 1) {
        if (tid < st) { s_sum[tid] += s_sum[tid + st]; s_sq[tid] += s_sq[tid + st]; }
        __syncthreads();
    }
    float mean = s_sum[0] * (1.0f / 1024.0f);
    float var  = (s_sq[0] - 1024.0f * mean * mean) * (1.0f / 1023.0f);  // ddof=1
    float rs   = gamma[0] / (sqrtf(fmaxf(var, 0.f)) + 1e-8f);
    float bt   = beta[0];
    unsigned short* orow = out + (size_t)row * 1024;
    for (int i = 0; i < 4; ++i)
        orow[tid + 256 * i] = f2bf_u((v[i] - mean) * rs + bt);
}

// ---------------------------------------------------------------------------
// GEMM: C[M,N] = A[M,K](bf16) @ W[N,K](bf16)^T + bias[N]
// Block = 256 threads = 8 waves (2x4), wave tile 32x32 -> block tile 64x128.
// A (64x32) and B (128x32) K-chunks staged in LDS via async global->LDS
// copies, double buffered; fragments then come from LDS (ds_load_b128).
// STORE_MODE: 0 = bf16 row-major (optionally ReLU)
//             1 = fp32, += resid[M,N]
//             2 = bf16 into [B,H,S,64]     (Q, K heads layout; H=16)
//             3 = bf16 into [B,H,64,S]     (V transposed for attention)
// ---------------------------------------------------------------------------
template <int STORE_MODE, bool RELU>
__global__ __launch_bounds__(256, 2)
void gemm_bf16_kernel(const unsigned short* __restrict__ A,
                      const unsigned short* __restrict__ W,
                      const float* __restrict__ bias,
                      const float* __restrict__ resid,
                      void* __restrict__ outp,
                      int M, int N, int K, int S) {
    __shared__ unsigned short sA[2][64 * 32];    // 8 KB
    __shared__ unsigned short sB[2][128 * 32];   // 16 KB
    const int tid  = threadIdx.x;
    const int lane = tid & 31;
    const int wave = tid >> 5;
    const int l15  = lane & 15;
    const int hi   = lane >> 4;            // half-wave select
    const int wm   = wave & 1;
    const int wn   = wave >> 1;
    const int bm   = blockIdx.y * 64;
    const int bn   = blockIdx.x * 128;

    const int aoff = hi * 8;               // A frag: K-offset per half-wave
    const int boff = hi * 8;               // B frag: K-offset per half-wave

    // Copy assignments: A tile 64x32 -> 16B/thread, B tile 128x32 -> 32B/thread.
    const int arow = tid >> 2;             // 0..63
    const int acol = (tid & 3) * 8;        // 0,8,16,24
    const int brow = tid >> 1;             // 0..127
    const int bcol = (tid & 1) * 16;       // 0,16
    const unsigned short* gA = A + (size_t)(bm + arow) * K + acol;
    const unsigned short* gW = W + (size_t)(bn + brow) * K + bcol;

    // Fragment read bases (within one LDS buffer; rows of 32 bf16 = 64 B).
    const int la0 = (wm * 32 + l15) * 32;
    const int lb0 = (wn * 32 + l15) * 32;

    f32x8 acc[2][2];
    for (int t = 0; t < 2; ++t)
        for (int u = 0; u < 2; ++u)
            for (int r = 0; r < 8; ++r) acc[t][u][r] = 0.f;

    const int nkc = K >> 5;

    auto issue_copy = [&](int kc) {
        const int buf = kc & 1;
        const int k0  = kc << 5;
        copy16_g2l(gA + k0,     &sA[buf][arow * 32 + acol]);
        copy16_g2l(gW + k0,     &sB[buf][brow * 32 + bcol]);
        copy16_g2l(gW + k0 + 8, &sB[buf][brow * 32 + bcol + 8]);
    };

    issue_copy(0);
    for (int kc = 0; kc < nkc; ++kc) {
        wait_async_copies();      // own chunk-kc copies have landed
        __syncthreads();          // ... and everyone else's are visible
        if (kc + 1 < nkc) issue_copy(kc + 1);   // overlap next chunk

        const unsigned short* tA = &sA[kc & 1][0];
        const unsigned short* tB = &sB[kc & 1][0];
        const unsigned short* A0 = tA + la0;
        const unsigned short* A1 = A0 + 16 * 32;
        const unsigned short* B0 = tB + lb0;
        const unsigned short* B1 = B0 + 16 * 32;

        BFrag a0, a1, b0, b1;
        a0.h[0] = load_bf8(A0 + aoff);
        a0.h[1] = load_bf8(A0 + 16 + aoff);
        a1.h[0] = load_bf8(A1 + aoff);
        a1.h[1] = load_bf8(A1 + 16 + aoff);
        b0.h[0] = load_bf8(B0 + boff);
        b0.h[1] = load_bf8(B0 + boff + 16);
        b1.h[0] = load_bf8(B1 + boff);
        b1.h[1] = load_bf8(B1 + boff + 16);
        acc[0][0] = wmma_bf16(a0, b0, acc[0][0]);
        acc[0][1] = wmma_bf16(a0, b1, acc[0][1]);
        acc[1][0] = wmma_bf16(a1, b0, acc[1][0]);
        acc[1][1] = wmma_bf16(a1, b1, acc[1][1]);
    }

    const float biasv[2] = { bias[bn + wn * 32 + l15], bias[bn + wn * 32 + 16 + l15] };

    for (int t = 0; t < 2; ++t) {
        for (int u = 0; u < 2; ++u) {
            for (int r = 0; r < 8; ++r) {
                int m = bm + wm * 32 + t * 16 + r + hi * 8;  // C layout: M = r + 8*hi
                int n = bn + wn * 32 + u * 16 + l15;         //           N = lane&15
                float val = acc[t][u][r] + biasv[u];
                if (RELU) val = fmaxf(val, 0.f);
                if (STORE_MODE == 0) {
                    ((unsigned short*)outp)[(size_t)m * N + n] = f2bf_u(val);
                } else if (STORE_MODE == 1) {
                    size_t idx = (size_t)m * N + n;
                    ((float*)outp)[idx] = resid[idx] + val;
                } else if (STORE_MODE == 2) {                // [B,H,S,64], H=16
                    int b = m / S, s = m - b * S;
                    int h = n >> 6, d = n & 63;
                    size_t idx = (((size_t)(b * 16 + h)) * S + s) * 64 + d;
                    ((unsigned short*)outp)[idx] = f2bf_u(val);
                } else {                                     // [B,H,64,S] (V^T)
                    int b = m / S, s = m - b * S;
                    int h = n >> 6, d = n & 63;
                    size_t idx = (((size_t)(b * 16 + h)) * 64 + d) * (size_t)S + s;
                    ((unsigned short*)outp)[idx] = f2bf_u(val);
                }
            }
        }
    }
}

// ---------------------------------------------------------------------------
// Flash attention: per wave one 16-query tile, dk = 64, keys in blocks of 32.
// scores = Q(16x64) @ K^T  -> 2x (2 WMMA);  ctx += P(16x32) @ V(32x64) -> 4 WMMA
// P relayout (C-layout -> A-layout) through per-wave LDS (in-order DS).
// Q,K in [B,H,S,64] bf16; Vt in [B,H,64,S] bf16; ctx out bf16 [B*S, 1024].
// ---------------------------------------------------------------------------
__global__ __launch_bounds__(256, 2)
void flash_attn_kernel(const unsigned short* __restrict__ Q,
                       const unsigned short* __restrict__ Km,
                       const unsigned short* __restrict__ Vt,
                       const int* __restrict__ mask,
                       unsigned short* __restrict__ ctx,
                       int S) {
    __shared__ unsigned short plds[8 * 16 * 32];      // 16x32 bf16 per wave
    const int lane = threadIdx.x & 31;
    const int wave = threadIdx.x >> 5;
    const int l15  = lane & 15;
    const int hi   = lane >> 4;
    const int bh   = blockIdx.x;                      // 0..B*H-1 (H=16)
    const int b    = bh >> 4;
    const int head = bh & 15;
    const int q0   = (blockIdx.y * 8 + wave) * 16;

    const unsigned short* Qb = Q  + ((size_t)bh * S + q0) * 64;
    const unsigned short* Kb = Km + (size_t)bh * S * 64;
    const unsigned short* Vb = Vt + (size_t)bh * 64 * S;
    const int* mrow = mask + (size_t)b * S;

    const int aoff = hi * 8;
    const int boff = hi * 8;

    BFrag qf[2];                                      // Q frags, d = 0..31 / 32..63
    for (int j = 0; j < 2; ++j) {
        qf[j].h[0] = load_bf8(Qb + (size_t)l15 * 64 + j * 32 + aoff);
        qf[j].h[1] = load_bf8(Qb + (size_t)l15 * 64 + j * 32 + 16 + aoff);
    }

    f32x8 acc[4];
    float mrun[8], lrun[8];
    for (int t = 0; t < 4; ++t)
        for (int r = 0; r < 8; ++r) acc[t][r] = 0.f;
    for (int r = 0; r < 8; ++r) { mrun[r] = -3.0e38f; lrun[r] = 0.f; }

    unsigned short* pbase = plds + wave * 512;

    for (int kb = 0; kb < S; kb += 32) {
        // ---- scores: two 16x16 tiles over 32 keys ----
        f32x8 s0, s1;
        for (int r = 0; r < 8; ++r) { s0[r] = 0.f; s1[r] = 0.f; }
        for (int dch = 0; dch < 2; ++dch) {
            BFrag k0f, k1f;
            const unsigned short* K0 = Kb + (size_t)(kb + l15) * 64 + dch * 32;
            const unsigned short* K1 = Kb + (size_t)(kb + 16 + l15) * 64 + dch * 32;
            k0f.h[0] = load_bf8(K0 + boff);
            k0f.h[1] = load_bf8(K0 + boff + 16);
            k1f.h[0] = load_bf8(K1 + boff);
            k1f.h[1] = load_bf8(K1 + boff + 16);
            s0 = wmma_bf16(qf[dch], k0f, s0);
            s1 = wmma_bf16(qf[dch], k1f, s1);
        }
        // scale 1/sqrt(64) + mask
        const int mk0 = mrow[kb + l15];
        const int mk1 = mrow[kb + 16 + l15];
        for (int r = 0; r < 8; ++r) {
            s0[r] = mk0 ? s0[r] * 0.125f : -1.0e9f;
            s1[r] = mk1 ? s1[r] * 0.125f : -1.0e9f;
        }
        // ---- online softmax (row spans 16 lanes of one half-wave) ----
        float mx[8];
        for (int r = 0; r < 8; ++r) mx[r] = fmaxf(s0[r], s1[r]);
        for (int off = 1; off < 16; off <<= 1)
            for (int r = 0; r < 8; ++r)
                mx[r] = fmaxf(mx[r], __shfl_xor(mx[r], off, 32));
        float alpha[8], rs[8];
        for (int r = 0; r < 8; ++r) {
            float mnew = fmaxf(mrun[r], mx[r]);
            alpha[r] = __expf(mrun[r] - mnew);
            mrun[r]  = mnew;
            s0[r] = __expf(s0[r] - mnew);
            s1[r] = __expf(s1[r] - mnew);
            rs[r] = s0[r] + s1[r];
        }
        for (int off = 1; off < 16; off <<= 1)
            for (int r = 0; r < 8; ++r)
                rs[r] += __shfl_xor(rs[r], off, 32);
        for (int r = 0; r < 8; ++r) lrun[r] = lrun[r] * alpha[r] + rs[r];
        for (int t = 0; t < 4; ++t)
            for (int r = 0; r < 8; ++r) acc[t][r] *= alpha[r];

        // ---- P: C-layout -> LDS row-major [16][32] -> A-layout frag ----
        for (int r = 0; r < 8; ++r) {
            int mr = r + hi * 8;
            pbase[mr * 32 + l15]      = f2bf_u(s0[r]);
            pbase[mr * 32 + 16 + l15] = f2bf_u(s1[r]);
        }
        BFrag pa;   // same-wave DS ops are in-order; compiler inserts s_wait_dscnt
        pa.h[0] = load_bf8(pbase + l15 * 32 + aoff);
        pa.h[1] = load_bf8(pbase + l15 * 32 + 16 + aoff);

        // ---- ctx += P @ V : four 16x16 d-tiles ----
        for (int t = 0; t < 4; ++t) {
            BFrag vf;
            const unsigned short* Vr = Vb + (size_t)(t * 16 + l15) * S + kb;
            vf.h[0] = load_bf8(Vr + boff);
            vf.h[1] = load_bf8(Vr + boff + 16);
            acc[t] = wmma_bf16(pa, vf, acc[t]);
        }
    }

    // ---- normalize and scatter back to [B*S, 1024] bf16 ----
    for (int r = 0; r < 8; ++r) {
        float inv = 1.0f / lrun[r];
        int s = q0 + r + hi * 8;
        size_t rowbase = ((size_t)b * S + s) * 1024 + head * 64;
        for (int t = 0; t < 4; ++t)
            ctx[rowbase + t * 16 + l15] = f2bf_u(acc[t][r] * inv);
    }
}

// ---------------------------------------------------------------------------
// Orchestration
// ---------------------------------------------------------------------------
extern "C" void kernel_launch(void* const* d_in, const int* in_sizes, int n_in,
                              void* d_out, int out_size, void* d_ws, size_t ws_size,
                              hipStream_t stream) {
    (void)in_sizes; (void)n_in; (void)out_size; (void)ws_size;

    const float* x    = (const float*)d_in[0];
    const int*   mask = (const int*)  d_in[1];
    const float* Wq   = (const float*)d_in[2];
    const float* bq   = (const float*)d_in[3];
    const float* Wk   = (const float*)d_in[4];
    const float* bk   = (const float*)d_in[5];
    const float* Wv   = (const float*)d_in[6];
    const float* bv   = (const float*)d_in[7];
    const float* Wo   = (const float*)d_in[8];
    const float* bo   = (const float*)d_in[9];
    const float* W1   = (const float*)d_in[10];
    const float* b1   = (const float*)d_in[11];
    const float* W2   = (const float*)d_in[12];
    const float* b2   = (const float*)d_in[13];
    const float* g1   = (const float*)d_in[14];
    const float* be1  = (const float*)d_in[15];
    const float* g2   = (const float*)d_in[16];
    const float* be2  = (const float*)d_in[17];

    constexpr int Bx = 2, Sx = 2048, D = 1024, FF = 4096;
    constexpr int M = Bx * Sx;     // 4096 tokens

    const size_t MB = 1ull << 20;
    unsigned char* ws = (unsigned char*)d_ws;
    unsigned short* hb   = (unsigned short*)(ws + 0);        // 8MB  LN out (reused for LN2)
    unsigned short* wqb  = (unsigned short*)(ws + 8  * MB);  // 2MB
    unsigned short* wkb  = (unsigned short*)(ws + 10 * MB);  // 2MB
    unsigned short* wvb  = (unsigned short*)(ws + 12 * MB);  // 2MB
    unsigned short* wob  = (unsigned short*)(ws + 14 * MB);  // 2MB
    unsigned short* w1b  = (unsigned short*)(ws + 16 * MB);  // 8MB
    unsigned short* w2b  = (unsigned short*)(ws + 24 * MB);  // 8MB
    unsigned short* qb   = (unsigned short*)(ws + 32 * MB);  // 8MB  [B,H,S,64]
    unsigned short* kbp  = (unsigned short*)(ws + 40 * MB);  // 8MB  [B,H,S,64]
    unsigned short* vtb  = (unsigned short*)(ws + 48 * MB);  // 8MB  [B,H,64,S]
    unsigned short* ctxb = (unsigned short*)(ws + 56 * MB);  // 8MB  [M,1024]
    unsigned short* f1b  = (unsigned short*)(ws + 32 * MB);  // 32MB reuse q/k/v/ctx
    float*          x2   = (float*)         (ws + 64 * MB);  // 16MB (total 80MB)

    // weights -> bf16
    cvt_f32_to_bf16_kernel<<<(D * D + 255) / 256, 256, 0, stream>>>(Wq, wqb, D * D);
    cvt_f32_to_bf16_kernel<<<(D * D + 255) / 256, 256, 0, stream>>>(Wk, wkb, D * D);
    cvt_f32_to_bf16_kernel<<<(D * D + 255) / 256, 256, 0, stream>>>(Wv, wvb, D * D);
    cvt_f32_to_bf16_kernel<<<(D * D + 255) / 256, 256, 0, stream>>>(Wo, wob, D * D);
    cvt_f32_to_bf16_kernel<<<(FF * D + 255) / 256, 256, 0, stream>>>(W1, w1b, FF * D);
    cvt_f32_to_bf16_kernel<<<(D * FF + 255) / 256, 256, 0, stream>>>(W2, w2b, D * FF);

    // pre-norm 1
    layernorm_bf16_kernel<<<M, 256, 0, stream>>>(x, g1, be1, hb);

    // QKV projections (WMMA GEMMs, async-LDS staged)
    dim3 gD(D / 128, M / 64);   // (8, 64)
    gemm_bf16_kernel<2, false><<<gD, 256, 0, stream>>>(hb, wqb, bq, nullptr, qb,  M, D, D, Sx);
    gemm_bf16_kernel<2, false><<<gD, 256, 0, stream>>>(hb, wkb, bk, nullptr, kbp, M, D, D, Sx);
    gemm_bf16_kernel<3, false><<<gD, 256, 0, stream>>>(hb, wvb, bv, nullptr, vtb, M, D, D, Sx);

    // attention
    flash_attn_kernel<<<dim3(Bx * 16, Sx / 128), 256, 0, stream>>>(qb, kbp, vtb, mask, ctxb, Sx);

    // output projection + residual (fp32)
    gemm_bf16_kernel<1, false><<<gD, 256, 0, stream>>>(ctxb, wob, bo, x, x2, M, D, D, Sx);

    // pre-norm 2
    layernorm_bf16_kernel<<<M, 256, 0, stream>>>(x2, g2, be2, hb);

    // FFN
    dim3 gF(FF / 128, M / 64);  // (32, 64)
    gemm_bf16_kernel<0, true ><<<gF, 256, 0, stream>>>(hb,  w1b, b1, nullptr, f1b, M, FF, D, Sx);
    gemm_bf16_kernel<1, false><<<gD, 256, 0, stream>>>(f1b, w2b, b2, x2, (float*)d_out, M, D, FF, Sx);
}
